// GPT_63419487092905
// MI455X (gfx1250) — matrix operand order repaired and checked by reading, not compiled, not run on hardware
//
#include <hip/hip_runtime.h>
#include <hip/hip_bf16.h>
#include <math.h>

typedef _Float16 half_t;
typedef __attribute__((ext_vector_type(16))) _Float16 v16h;
typedef __attribute__((ext_vector_type(8)))  _Float16 v8h;
typedef __attribute__((ext_vector_type(8)))  float    v8f;
typedef __attribute__((ext_vector_type(4)))  float    v4f;
typedef int v4i __attribute__((vector_size(16)));   // matches builtin param type

#define S_LEN  2048
#define EDIM   1024
#define HN     16
#define HD     64
#define NLAYER 6
#define FDIM   4096
#define VOCAB  32000

#if defined(__has_builtin)
#if __has_builtin(__builtin_amdgcn_global_load_async_to_lds_b128) && \
    __has_builtin(__builtin_amdgcn_s_wait_asynccnt)
#define HAVE_ASYNC 1
#else
#define HAVE_ASYNC 0
#endif
#else
#define HAVE_ASYNC 0
#endif

__device__ __forceinline__ v16h cat16(v8h lo, v8h hi) {
  return __builtin_shufflevector(lo, hi, 0,1,2,3,4,5,6,7,8,9,10,11,12,13,14,15);
}

// ---------------- embedding: x = W_emb[tok] + W_pos ----------------
__global__ void embed_kernel(const int* __restrict__ tok, const float* __restrict__ Wemb,
                             const float* __restrict__ Wpos, float* __restrict__ x) {
  int idx = blockIdx.x * blockDim.x + threadIdx.x;   // S*E
  int s = idx >> 10;
  int e = idx & (EDIM - 1);
  int t = tok[s];
  x[idx] = Wemb[(size_t)t * EDIM + e] + Wpos[idx];
}

// ---------------- layernorm: h_f16 = LN(x)*g + b ----------------
__global__ __launch_bounds__(256)
void ln_kernel(const float* __restrict__ x, const float* __restrict__ g,
               const float* __restrict__ b, half_t* __restrict__ h) {
  __shared__ float red[256];
  int row = blockIdx.x, tid = threadIdx.x;
  const float* xr = x + (size_t)row * EDIM;
  float v0 = xr[tid], v1 = xr[tid + 256], v2 = xr[tid + 512], v3 = xr[tid + 768];
  red[tid] = v0 + v1 + v2 + v3;
  __syncthreads();
  for (int off = 128; off > 0; off >>= 1) { if (tid < off) red[tid] += red[tid + off]; __syncthreads(); }
  float mu = red[0] * (1.0f / EDIM);
  __syncthreads();
  float d0 = v0 - mu, d1 = v1 - mu, d2 = v2 - mu, d3 = v3 - mu;
  red[tid] = d0 * d0 + d1 * d1 + d2 * d2 + d3 * d3;
  __syncthreads();
  for (int off = 128; off > 0; off >>= 1) { if (tid < off) red[tid] += red[tid + off]; __syncthreads(); }
  float rstd = rsqrtf(red[0] * (1.0f / EDIM) + 1e-5f);
  half_t* hr = h + (size_t)row * EDIM;
  hr[tid]       = (half_t)(d0 * rstd * g[tid]       + b[tid]);
  hr[tid + 256] = (half_t)(d1 * rstd * g[tid + 256] + b[tid + 256]);
  hr[tid + 512] = (half_t)(d2 * rstd * g[tid + 512] + b[tid + 512]);
  hr[tid + 768] = (half_t)(d3 * rstd * g[tid + 768] + b[tid + 768]);
}

// ---------------- transpose V: [S,E] -> [E,S] (f16) ----------------
__global__ void transpose_kernel(const half_t* __restrict__ v, half_t* __restrict__ vT) {
  int idx = blockIdx.x * blockDim.x + threadIdx.x;   // S*E
  int s = idx >> 10;
  int c = idx & (EDIM - 1);
  vT[(size_t)c * S_LEN + s] = v[idx];
}

// ---------------- WMMA GEMM: C[M,N] = A[M,K](f16) * W[N,K]^T ----------------
// LDS-staged, double-buffered. A tile copied via async-to-LDS (f16, raw bytes);
// B tile converted f32->f16 cooperatively (once per block) with packed cvt.
// EPI: 0 = store f16 ; 1 = gelu(bias+acc) f16 ; 2 = resid + acc + bias f32 ; 3 = store f32
template<int EPI>
__global__ __launch_bounds__(256)
void gemm_kernel(const half_t* __restrict__ A, const float* __restrict__ W,
                 const float* __restrict__ bias, const float* __restrict__ resid,
                 void* __restrict__ outv, int M, int N, int K) {
  (void)M;
  __shared__ __align__(16) half_t At[2][128 * 32];
  __shared__ __align__(16) half_t Bt[2][128 * 32];

  const int tid   = threadIdx.x;
  const int lane  = tid & 31;
  const int wave  = tid >> 5;
  const int waveM = wave & 3;            // 4 waves along M
  const int waveN = wave >> 2;           // 2 waves along N
  const int mwave = waveM * 32;
  const int nwave = waveN * 64;
  const int mbase = blockIdx.y * 128;
  const int nbase = blockIdx.x * 128;
  const int mlo = lane & 15, hi = lane >> 4;

  // staging assignment: thread -> (row 0..127, 16-element half)
  const int srow  = tid >> 1;
  const int shalf = tid & 1;
  const half_t* agbl = A + (size_t)(mbase + srow) * K + shalf * 16;
  const float*  wgbl = W + (size_t)(nbase + srow) * K + shalf * 16;

  auto stage_a = [&](int p, int kk) {
    const half_t* g = agbl + kk;
    half_t* l = &At[p][srow * 32 + shalf * 16];
#if HAVE_ASYNC
    __builtin_amdgcn_global_load_async_to_lds_b128((v4i*)(void*)(const void*)g,
                                                   (v4i*)(void*)l, 0, 0);
    __builtin_amdgcn_global_load_async_to_lds_b128((v4i*)(void*)(const void*)(g + 8),
                                                   (v4i*)(void*)(l + 8), 0, 0);
#else
    *(v8h*)l       = *(const v8h*)g;
    *(v8h*)(l + 8) = *(const v8h*)(g + 8);
#endif
  };
  auto load_b = [&](int kk, v4f (&wv)[4]) {
    const float* g = wgbl + kk;
    wv[0] = *(const v4f*)(g);
    wv[1] = *(const v4f*)(g + 4);
    wv[2] = *(const v4f*)(g + 8);
    wv[3] = *(const v4f*)(g + 12);
  };
  auto store_b = [&](int p, v4f (&wv)[4]) {
    v16h bb;
#pragma unroll
    for (int j = 0; j < 8; j++) {
      auto t = __builtin_amdgcn_cvt_pkrtz(wv[j >> 1][(j & 1) * 2], wv[j >> 1][(j & 1) * 2 + 1]);
      bb[2 * j]     = (half_t)t[0];
      bb[2 * j + 1] = (half_t)t[1];
    }
    *(v16h*)&Bt[p][srow * 32 + shalf * 16] = bb;
  };

  v8f zero = {0.f,0.f,0.f,0.f,0.f,0.f,0.f,0.f};
  v8f acc[2][4];
#pragma unroll
  for (int i = 0; i < 2; i++)
#pragma unroll
    for (int j = 0; j < 4; j++) acc[i][j] = zero;

  // prologue: stage k-tile 0
  {
    stage_a(0, 0);
    v4f w0[4];
    load_b(0, w0);
    store_b(0, w0);
  }
#if HAVE_ASYNC
  __builtin_amdgcn_s_wait_asynccnt(0);
#endif
  __syncthreads();

  int p = 0;
  for (int kk = 0; kk < K; kk += 32) {
    v4f wn[4];
    const bool nxt = (kk + 32) < K;
    if (nxt) {             // issue next-tile copies before computing (overlap)
      stage_a(p ^ 1, kk + 32);
      load_b(kk + 32, wn);
    }
    // 8 WMMAs on current tile from LDS
    v16h a[2];
#pragma unroll
    for (int ms = 0; ms < 2; ms++) {
      const half_t* ap = &At[p][(mwave + ms * 16 + mlo) * 32];
      a[ms] = cat16(*(const v8h*)(ap + hi * 16), *(const v8h*)(ap + hi * 16 + 8));
    }
#pragma unroll
    for (int ns = 0; ns < 4; ns++) {
      const half_t* bp = &Bt[p][(nwave + ns * 16 + mlo) * 32 + hi * 16];
      v16h bq = cat16(*(const v8h*)(bp), *(const v8h*)(bp + 8));
#pragma unroll
      for (int ms = 0; ms < 2; ms++)
        acc[ms][ns] = __builtin_amdgcn_wmma_f32_16x16x32_f16(
            false, a[ms], false, bq, (short)0, acc[ms][ns], false, false);
    }
    if (nxt) {
      store_b(p ^ 1, wn);
#if HAVE_ASYNC
      __builtin_amdgcn_s_wait_asynccnt(0);
#endif
    }
    __syncthreads();
    p ^= 1;
  }

#pragma unroll
  for (int ms = 0; ms < 2; ms++) {
#pragma unroll
    for (int ns = 0; ns < 4; ns++) {
      int c = nbase + nwave + ns * 16 + mlo;
#pragma unroll
      for (int i = 0; i < 8; i++) {
        int r = mbase + mwave + ms * 16 + hi * 8 + i;
        size_t off = (size_t)r * N + c;
        float vv = acc[ms][ns][i];
        if constexpr (EPI == 0) {
          ((half_t*)outv)[off] = (half_t)vv;
        } else if constexpr (EPI == 1) {
          float t  = vv + bias[c];
          float ge = 0.5f * t * (1.0f + erff(t * 0.70710678118654752f));
          ((half_t*)outv)[off] = (half_t)ge;
        } else if constexpr (EPI == 2) {
          ((float*)outv)[off] = resid[off] + vv + bias[c];
        } else {
          ((float*)outv)[off] = vv;
        }
      }
    }
  }
}

// ---------------- flash attention: 1 wave per (16 q rows, head) ----------------
__global__ __launch_bounds__(32)
void attn_kernel(const half_t* __restrict__ q, const half_t* __restrict__ k,
                 const half_t* __restrict__ vT, half_t* __restrict__ outp) {
  __shared__ __align__(16) half_t P[16 * 32];
  const int lane = threadIdx.x;
  const int mlo = lane & 15, hi = lane >> 4;
  const int qb = blockIdx.x * 16;
  const int head = blockIdx.y;
  const float scale = 0.125f;   // 1/sqrt(64)

  v16h qa[2];
  {
    const half_t* qp = q + (size_t)(qb + mlo) * EDIM + head * HD;
#pragma unroll
    for (int ks = 0; ks < 2; ks++) {
      v8h lo = *(const v8h*)(qp + ks * 32 + hi * 16);
      v8h hh = *(const v8h*)(qp + ks * 32 + hi * 16 + 8);
      qa[ks] = cat16(lo, hh);
    }
  }

  float mrow[8], lrow[8];
  v8f vzero = {0.f,0.f,0.f,0.f,0.f,0.f,0.f,0.f};
  v8f o[4];
#pragma unroll
  for (int i = 0; i < 8; i++) { mrow[i] = -1e30f; lrow[i] = 0.0f; }
#pragma unroll
  for (int t = 0; t < 4; t++) o[t] = vzero;

  const int nblk = (qb + 16 + 31) >> 5;
  for (int b = 0; b < nblk; b++) {
    const int kv = b * 32;
    v8f c0 = vzero, c1 = vzero;
    {
      const half_t* kp0 = k + (size_t)(kv + mlo) * EDIM + head * HD;
      const half_t* kp1 = k + (size_t)(kv + 16 + mlo) * EDIM + head * HD;
#pragma unroll
      for (int ks = 0; ks < 2; ks++) {
        v16h kb0 = cat16(*(const v8h*)(kp0 + ks * 32 + hi * 16),
                         *(const v8h*)(kp0 + ks * 32 + hi * 16 + 8));
        v16h kb1 = cat16(*(const v8h*)(kp1 + ks * 32 + hi * 16),
                         *(const v8h*)(kp1 + ks * 32 + hi * 16 + 8));
        c0 = __builtin_amdgcn_wmma_f32_16x16x32_f16(false, qa[ks], false, kb0, (short)0, c0, false, false);
        c1 = __builtin_amdgcn_wmma_f32_16x16x32_f16(false, qa[ks], false, kb1, (short)0, c1, false, false);
      }
    }
    float s0[8], s1[8];
#pragma unroll
    for (int i = 0; i < 8; i++) {
      int row = qb + hi * 8 + i;
      s0[i] = ((kv + mlo)      <= row) ? c0[i] * scale : -1e30f;
      s1[i] = ((kv + 16 + mlo) <= row) ? c1[i] * scale : -1e30f;
    }
    float nm[8], al[8];
#pragma unroll
    for (int i = 0; i < 8; i++) {
      float mx = fmaxf(s0[i], s1[i]);
      mx = fmaxf(mx, __shfl_xor(mx, 1));
      mx = fmaxf(mx, __shfl_xor(mx, 2));
      mx = fmaxf(mx, __shfl_xor(mx, 4));
      mx = fmaxf(mx, __shfl_xor(mx, 8));
      nm[i] = fmaxf(mrow[i], mx);
      al[i] = __expf(mrow[i] - nm[i]);
      mrow[i] = nm[i];
    }
#pragma unroll
    for (int i = 0; i < 8; i++) {
      float p0 = __expf(s0[i] - nm[i]);
      float p1 = __expf(s1[i] - nm[i]);
      P[(hi * 8 + i) * 32 + mlo]      = (half_t)p0;
      P[(hi * 8 + i) * 32 + 16 + mlo] = (half_t)p1;
      float rs = p0 + p1;
      rs += __shfl_xor(rs, 1);
      rs += __shfl_xor(rs, 2);
      rs += __shfl_xor(rs, 4);
      rs += __shfl_xor(rs, 8);
      lrow[i] = lrow[i] * al[i] + rs;
    }
    __syncthreads();   // single-wave block: dscnt wait + nop barrier
    v16h pa;
    {
      const half_t* pp = P + mlo * 32;
      pa = cat16(*(const v8h*)(pp + hi * 16), *(const v8h*)(pp + hi * 16 + 8));
    }
#pragma unroll
    for (int t = 0; t < 4; t++) {
      v8f ot = o[t];
#pragma unroll
      for (int i = 0; i < 8; i++) ot[i] *= al[i];
      const half_t* vp = vT + (size_t)(head * HD + t * 16 + mlo) * S_LEN + kv + hi * 16;
      v16h vb = cat16(*(const v8h*)(vp), *(const v8h*)(vp + 8));
      o[t] = __builtin_amdgcn_wmma_f32_16x16x32_f16(false, pa, false, vb, (short)0, ot, false, false);
    }
    __syncthreads();
  }
#pragma unroll
  for (int t = 0; t < 4; t++) {
#pragma unroll
    for (int i = 0; i < 8; i++) {
      int row = qb + hi * 8 + i;
      int col = head * HD + t * 16 + mlo;
      outp[(size_t)row * EDIM + col] = (half_t)(o[t][i] / lrow[i]);
    }
  }
}

// ---------------- driver ----------------
extern "C" void kernel_launch(void* const* d_in, const int* in_sizes, int n_in,
                              void* d_out, int out_size, void* d_ws, size_t ws_size,
                              hipStream_t stream) {
  (void)in_sizes; (void)n_in; (void)out_size; (void)ws_size;
  const int*   tok  = (const int*)d_in[0];
  const float* Wemb = (const float*)d_in[1];
  const float* Wpos = (const float*)d_in[2];
  const float* Wq   = (const float*)d_in[3];
  const float* Wk   = (const float*)d_in[4];
  const float* Wv   = (const float*)d_in[5];
  const float* Wo   = (const float*)d_in[6];
  const float* bo   = (const float*)d_in[7];
  const float* ln1g = (const float*)d_in[8];
  const float* ln1b = (const float*)d_in[9];
  const float* ln2g = (const float*)d_in[10];
  const float* ln2b = (const float*)d_in[11];
  const float* W1   = (const float*)d_in[12];
  const float* b1   = (const float*)d_in[13];
  const float* W2   = (const float*)d_in[14];
  const float* b2   = (const float*)d_in[15];
  const float* lnfg = (const float*)d_in[16];
  const float* lnfb = (const float*)d_in[17];

  char* ws = (char*)d_ws;
  size_t off = 0;
  float*  x   = (float*)(ws + off);  off += (size_t)S_LEN * EDIM * 4;
  half_t* h   = (half_t*)(ws + off); off += (size_t)S_LEN * EDIM * 2;
  half_t* qf  = (half_t*)(ws + off); off += (size_t)S_LEN * EDIM * 2;
  half_t* kf  = (half_t*)(ws + off); off += (size_t)S_LEN * EDIM * 2;
  half_t* vf  = (half_t*)(ws + off); off += (size_t)S_LEN * EDIM * 2;
  half_t* vTf = (half_t*)(ws + off); off += (size_t)S_LEN * EDIM * 2;
  half_t* att = (half_t*)(ws + off); off += (size_t)S_LEN * EDIM * 2;
  half_t* ff1 = (half_t*)(ws + off); off += (size_t)S_LEN * FDIM * 2;

  const int nElem = S_LEN * EDIM;
  embed_kernel<<<nElem / 256, 256, 0, stream>>>(tok, Wemb, Wpos, x);

  for (int l = 0; l < NLAYER; l++) {
    ln_kernel<<<S_LEN, 256, 0, stream>>>(x, ln1g + l * EDIM, ln1b + l * EDIM, h);
    gemm_kernel<0><<<dim3(EDIM / 128, S_LEN / 128), 256, 0, stream>>>(
        h, Wq + (size_t)l * EDIM * EDIM, nullptr, nullptr, qf, S_LEN, EDIM, EDIM);
    gemm_kernel<0><<<dim3(EDIM / 128, S_LEN / 128), 256, 0, stream>>>(
        h, Wk + (size_t)l * EDIM * EDIM, nullptr, nullptr, kf, S_LEN, EDIM, EDIM);
    gemm_kernel<0><<<dim3(EDIM / 128, S_LEN / 128), 256, 0, stream>>>(
        h, Wv + (size_t)l * EDIM * EDIM, nullptr, nullptr, vf, S_LEN, EDIM, EDIM);
    transpose_kernel<<<nElem / 256, 256, 0, stream>>>(vf, vTf);
    attn_kernel<<<dim3(S_LEN / 16, HN), 32, 0, stream>>>(qf, kf, vTf, att);
    gemm_kernel<2><<<dim3(EDIM / 128, S_LEN / 128), 256, 0, stream>>>(
        att, Wo + (size_t)l * EDIM * EDIM, bo + l * EDIM, x, x, S_LEN, EDIM, EDIM);
    ln_kernel<<<S_LEN, 256, 0, stream>>>(x, ln2g + l * EDIM, ln2b + l * EDIM, h);
    gemm_kernel<1><<<dim3(FDIM / 128, S_LEN / 128), 256, 0, stream>>>(
        h, W1 + (size_t)l * FDIM * EDIM, b1 + l * FDIM, nullptr, ff1, S_LEN, FDIM, EDIM);
    gemm_kernel<2><<<dim3(EDIM / 128, S_LEN / 128), 256, 0, stream>>>(
        ff1, W2 + (size_t)l * EDIM * FDIM, b2 + l * EDIM, x, x, S_LEN, EDIM, FDIM);
  }

  ln_kernel<<<S_LEN, 256, 0, stream>>>(x, lnfg, lnfb, h);
  gemm_kernel<3><<<dim3(VOCAB / 128, S_LEN / 128), 256, 0, stream>>>(
      h, Wemb, nullptr, nullptr, d_out, S_LEN, VOCAB, EDIM);
}
